// Encoder_for_PP_43078521979558
// MI455X (gfx1250) — matrix operand order repaired and checked by reading, not compiled, not run on hardware
//
#include <hip/hip_runtime.h>
#include <hip/hip_bf16.h>
#include <stdint.h>

// ---------------------------------------------------------------------------
// CDNA5 (gfx1250) fused graph-transformer encoder step, wave32 + WMMA bf16
// + Tensor Data Mover staging of LDS tiles.
// ---------------------------------------------------------------------------

typedef __attribute__((ext_vector_type(16))) __bf16 v16bf;
typedef __attribute__((ext_vector_type(8)))  float  v8f;
typedef unsigned int u32x4 __attribute__((ext_vector_type(4)));
typedef int          i32x4 __attribute__((ext_vector_type(4)));
typedef int          i32x8 __attribute__((ext_vector_type(8)));

union Frag16 { unsigned int u[8]; uint4 q[2]; v16bf v; };

#define NATOMS 200000
#define AF     151      // atom feature dim
#define BFD    165      // bond feature dim
#define MAXNB  6
#define HID    128
#define KA     160      // padded K for input projection (151 -> 160)
#define KM     320      // padded K for msg GEMM (293 -> 320)
#define KO     256      // K for output projection (2 heads * 128)

// workspace layout (bytes)
static const size_t OFF_H0  = 0;                                   // N*128 f32
static const size_t OFF_MSG = OFF_H0 + (size_t)NATOMS * HID * 4;   // N*320 bf16
static const size_t OFF_WI  = OFF_MSG + (size_t)NATOMS * KM * 2;   // 5*8 frags  * 1KB
static const size_t OFF_WH  = OFF_WI + (size_t)5 * 8 * 1024;       // 10*48 frags * 1KB
static const size_t OFF_WO  = OFF_WH + (size_t)10 * 48 * 1024;     // 8*8 frags  * 1KB

#if defined(__has_builtin)
#if __has_builtin(__builtin_amdgcn_tensor_load_to_lds)
#define USE_TDM 1
#endif
#endif
#ifndef USE_TDM
#define USE_TDM 0
#endif

__device__ __forceinline__ unsigned short f32_to_bf16(float f) {
    unsigned int u = __float_as_uint(f);
    unsigned int r = (u + 0x7FFFu + ((u >> 16) & 1u)) >> 16;   // RNE
    return (unsigned short)r;
}
__device__ __forceinline__ float bf2f(unsigned short h) {
    return __uint_as_float(((unsigned int)h) << 16);
}
__device__ __forceinline__ unsigned int pack_bf16(float a, float b) {
    return (unsigned int)f32_to_bf16(a) | ((unsigned int)f32_to_bf16(b) << 16);
}

#if USE_TDM
// 32-bit LDS byte offset of a __shared__ object (ptrtoint of addrspace(3) ptr)
__device__ __forceinline__ unsigned lds_addr_of(const void* p) {
    return (unsigned)(unsigned long long)
        (__attribute__((address_space(3))) const unsigned char*)(const unsigned char*)p;
}

// 1-D contiguous TDM copy: nelem4 4-byte elements, global -> LDS.
// D# per CDNA5 ISA (groups 2/3 zero => <=2D tensor mode).
__device__ __forceinline__ void tdm_copy_1d(unsigned lds_off, const void* gptr,
                                            unsigned nelem4) {
    unsigned long long ga = (unsigned long long)gptr;
    u32x4 g0;
    g0.x = 0x1u;                                            // count=1, user mode
    g0.y = lds_off;                                         // lds_addr (bytes)
    g0.z = (unsigned)(ga & 0xFFFFFFFFu);                    // global_addr[31:0]
    g0.w = (unsigned)((ga >> 32) & 0x01FFFFFFu) | (2u << 30); // addr[56:32] | type=2
    i32x8 g1;
    g1[0] = (int)(2u << 16);                                // data_size = 4B
    g1[1] = (int)((nelem4 & 0xFFFFu) << 16);                // tensor_dim0[15:0]
    g1[2] = (int)((nelem4 >> 16) & 0xFFFFu) | (int)(1u << 16); // td0[31:16] | tensor_dim1=1
    g1[3] = (int)((nelem4 & 0xFFFFu) << 16);                // tile_dim0
    g1[4] = 1;                                              // tile_dim1=1, tile_dim2=0
    g1[5] = (int)nelem4;                                    // tensor_dim0_stride[31:0]
    g1[6] = 0;
    g1[7] = 0;
    i32x4 z4 = {0, 0, 0, 0};
#if __clang_major__ >= 23
    i32x8 z8 = {0, 0, 0, 0, 0, 0, 0, 0};
    __builtin_amdgcn_tensor_load_to_lds(g0, g1, z4, z4, z8, 0);
#else
    __builtin_amdgcn_tensor_load_to_lds(g0, g1, z4, z4, 0);
#endif
}
#endif // USE_TDM

// ---------------------------------------------------------------------------
// Weight conversion into WMMA B-fragment layout:
//   frag(ks,nt) = 1024B: lane l owns words [l*8 .. l*8+7]; word i holds
//   bf16 pair (k, k+1) with k = ks*32 + (l>>4)*16 + 2i, col n = nt*16 + (l&15)
// ---------------------------------------------------------------------------
__global__ void cvt_plain(const float* __restrict__ W, unsigned int* __restrict__ dst,
                          int Ksrc, int ntiles, int total) {
    int w = blockIdx.x * blockDim.x + threadIdx.x;
    if (w >= total) return;
    int frag = w >> 8, rem = w & 255, lane = rem >> 3, i = rem & 7;
    int ks = frag / ntiles, nt = frag - ks * ntiles;
    int n  = nt * 16 + (lane & 15);
    int k  = ks * 32 + ((lane >> 4) * 16) + 2 * i;
    float v0 = (k     < Ksrc) ? W[(size_t)k       * HID + n] : 0.f;
    float v1 = (k + 1 < Ksrc) ? W[(size_t)(k + 1) * HID + n] : 0.f;
    dst[w] = pack_bf16(v0, v1);
}

// combined QKV weights: B is [320, 768]; col -> {branch q/k/v, head e, h}
__global__ void cvt_wh(const float* __restrict__ Wq, const float* __restrict__ Wk,
                       const float* __restrict__ Wv, unsigned int* __restrict__ dst,
                       int total) {
    int w = blockIdx.x * blockDim.x + threadIdx.x;
    if (w >= total) return;
    int frag = w >> 8, rem = w & 255, lane = rem >> 3, i = rem & 7;
    int ks = frag / 48, nt = frag - ks * 48;
    int col = nt * 16 + (lane & 15);
    const float* W = (col < 256) ? Wq : (col < 512) ? Wk : Wv;
    int e = (col >> 7) & 1, h = col & 127;
    int k = ks * 32 + ((lane >> 4) * 16) + 2 * i;
    float v0 = (k     < HID + BFD) ? W[((size_t)e * (HID + BFD) + k)     * HID + h] : 0.f;
    float v1 = (k + 1 < HID + BFD) ? W[((size_t)e * (HID + BFD) + k + 1) * HID + h] : 0.f;
    dst[w] = pack_bf16(v0, v1);
}

// ---------------------------------------------------------------------------
// k1: h0 = LN(relu(f_atoms @ W_i + b_i)).  One wave per 16-atom tile.
// ---------------------------------------------------------------------------
__global__ __launch_bounds__(32) void k1_inproj(
    const float* __restrict__ fA, const uint4* __restrict__ wiFrag,
    const float* __restrict__ bi, const float* __restrict__ g1,
    const float* __restrict__ b1, float* __restrict__ h0) {
    __shared__ __align__(16) unsigned char smraw[16 * HID * 4];      // 8192 B
    unsigned short* smA = (unsigned short*)smraw;                    // [16][160] bf16 (phase 1)
    float*          smH = (float*)smraw;                             // [16][128] f32  (phase 2)

    const int l = threadIdx.x, tile = blockIdx.x;
    const int m = l & 15, kb = (l >> 4) * 8;

    for (int idx = l; idx < 16 * KA; idx += 32) {
        int r = idx / KA, k = idx - r * KA;
        float v = (k < AF) ? fA[(size_t)(tile * 16 + r) * AF + k] : 0.f;
        smA[r * KA + k] = f32_to_bf16(v);
    }
    __syncthreads();

    Frag16 a[5];
#pragma unroll
    for (int ks = 0; ks < 5; ++ks)
#pragma unroll
        for (int i = 0; i < 8; ++i) {
            int k = ks * 32 + 2 * kb + 2 * i;
            a[ks].u[i] = *(const unsigned int*)&smA[m * KA + k];
        }
    __syncthreads();

#pragma unroll 1
    for (int nt = 0; nt < 8; ++nt) {
        v8f acc = {0.f, 0.f, 0.f, 0.f, 0.f, 0.f, 0.f, 0.f};
#pragma unroll
        for (int ks = 0; ks < 5; ++ks) {
            Frag16 bf;
            const uint4* p = wiFrag + (size_t)(ks * 8 + nt) * 64;
            bf.q[0] = p[2 * l];
            bf.q[1] = p[2 * l + 1];
            acc = __builtin_amdgcn_wmma_f32_16x16x32_bf16(false, a[ks].v, false, bf.v,
                                                          (short)0, acc, false, false);
        }
        int col = nt * 16 + m;
        float bias = bi[col];
#pragma unroll
        for (int r = 0; r < 8; ++r) {
            int M = (l >> 4) * 8 + r;
            float v = acc[r] + bias;
            smH[M * HID + col] = v > 0.f ? v : 0.f;
        }
    }
    __syncthreads();

    if (l < 16) {
        float s = 0.f, ss = 0.f;
        for (int c = 0; c < HID; ++c) { float v = smH[l * HID + c]; s += v; ss += v * v; }
        float mu  = s * (1.f / HID);
        float var = ss * (1.f / HID) - mu * mu;
        float inv = rsqrtf(var + 1e-5f);
        size_t base = (size_t)(tile * 16 + l) * HID;
        for (int c = 0; c < HID; ++c) {
            float v = smH[l * HID + c];
            h0[base + c] = g1[c] * (v - mu) * inv + b1[c];
        }
    }
}

// ---------------------------------------------------------------------------
// k2: msg[n] = sum_j [ h0[a2a[n,j]] , f_bonds[a2b[n,j]] ]  -> bf16 [N,320]
// ---------------------------------------------------------------------------
__global__ __launch_bounds__(128) void k2_gather(
    const float* __restrict__ h0, const float* __restrict__ fB,
    const int* __restrict__ a2a, const int* __restrict__ a2b,
    unsigned short* __restrict__ msg) {
    __shared__ int nb[12];
    int n = blockIdx.x, t = threadIdx.x;
    if (t < 6)       nb[t] = a2a[n * 6 + t];
    else if (t < 12) nb[t] = a2b[n * 6 + t - 6];
    __syncthreads();
#pragma unroll 1
    for (int k = t; k < KM; k += 128) {
        float s = 0.f;
        if (k < HID) {
#pragma unroll
            for (int j = 0; j < 6; ++j) s += h0[(size_t)nb[j] * HID + k];
        } else if (k < HID + BFD) {
            int kb = k - HID;
#pragma unroll
            for (int j = 0; j < 6; ++j) s += fB[(size_t)nb[6 + j] * BFD + kb];
        }
        msg[(size_t)n * KM + k] = f32_to_bf16(s);
    }
}

// ---------------------------------------------------------------------------
// k3: QKV GEMM (+h0+bias+relu) -> 2-head attention -> W_o GEMM -> LN -> +h0
// One wave per 16-atom tile; intermediates never leave LDS.  Tile staging via
// Tensor Data Mover (both tiles are contiguous row-blocks -> 1-D descriptors).
// ---------------------------------------------------------------------------
__global__ __launch_bounds__(32) void k3_fused(
    const unsigned short* __restrict__ msg, const float* __restrict__ h0,
    const uint4* __restrict__ whFrag, const uint4* __restrict__ woFrag,
    const float* __restrict__ bq, const float* __restrict__ bk, const float* __restrict__ bv,
    const float* __restrict__ bo, const float* __restrict__ g2, const float* __restrict__ b2,
    float* __restrict__ out) {
    __shared__ __align__(16) unsigned char smraw[40960];
    float*          smH  = (float*)smraw;                              // [16][128] f32, whole kernel
    unsigned short* smA  = (unsigned short*)(smraw + 8192);            // [16][320] bf16 (phase 1)
    unsigned short* smQ  = (unsigned short*)(smraw + 8192);            // [16][768] bf16 (phase 2+)
    unsigned short* smX  = (unsigned short*)(smraw + 8192 + 24576);    // [16][256] bf16 (phase 3)
    float*          smXO = (float*)(smraw + 8192 + 24576);             // [16][128] f32  (phase 4)

    const int l = threadIdx.x, tile = blockIdx.x;
    const int m = l & 15, kb = (l >> 4) * 8;

#if USE_TDM
    // async DMA: h0 tile (8192B) and msg tile (10240B), both contiguous
    tdm_copy_1d(lds_addr_of(smH), h0 + (size_t)tile * 16 * HID, 16 * HID);
    tdm_copy_1d(lds_addr_of(smA), msg + (size_t)tile * 16 * KM, 16 * KM / 2);
    __builtin_amdgcn_s_wait_tensorcnt(0);
#else
    for (int idx = l; idx < 16 * HID; idx += 32)
        smH[idx] = h0[(size_t)tile * 16 * HID + idx];
    for (int idx = l; idx < 16 * KM; idx += 32)
        smA[idx] = msg[(size_t)tile * 16 * KM + idx];
#endif
    __syncthreads();

    // build 10 A fragments (msg tile, K = 320)
    Frag16 a[10];
#pragma unroll
    for (int ks = 0; ks < 10; ++ks)
#pragma unroll
        for (int i = 0; i < 8; ++i) {
            int k = ks * 32 + 2 * kb + 2 * i;
            a[ks].u[i] = *(const unsigned int*)&smA[m * KM + k];
        }
    __syncthreads();

    // ---- QKV GEMM: [16,320] x [320,768], epilogue h0 + bias + relu -> LDS bf16
#pragma unroll 1
    for (int nt = 0; nt < 48; ++nt) {
        if (nt + 1 < 48)
            __builtin_prefetch((const void*)(whFrag + (size_t)(nt + 1) * 64), 0, 0);
        v8f acc = {0.f, 0.f, 0.f, 0.f, 0.f, 0.f, 0.f, 0.f};
#pragma unroll
        for (int ks = 0; ks < 10; ++ks) {
            Frag16 bf;
            const uint4* p = whFrag + (size_t)(ks * 48 + nt) * 64;
            bf.q[0] = p[2 * l];
            bf.q[1] = p[2 * l + 1];
            acc = __builtin_amdgcn_wmma_f32_16x16x32_bf16(false, a[ks].v, false, bf.v,
                                                          (short)0, acc, false, false);
        }
        int col = nt * 16 + m;
        const float* bptr = (col < 256) ? bq : (col < 512) ? bk : bv;
        int e = (col >> 7) & 1, h = col & 127;
        float bias = bptr[e * HID + h];
#pragma unroll
        for (int r = 0; r < 8; ++r) {
            int M = (l >> 4) * 8 + r;
            float v = acc[r] + smH[M * HID + h] + bias;
            smQ[M * 768 + col] = f32_to_bf16(v > 0.f ? v : 0.f);
        }
    }
    __syncthreads();

    // ---- attention: lane -> (atom = m, q-head = l>>4); heads axis is 2 wide
    {
        int q = l >> 4;
        const unsigned short* row = smQ + m * 768;
        float s0 = 0.f, s1 = 0.f;
        for (int h = 0; h < HID; ++h) {
            float qv = bf2f(row[q * HID + h]);
            s0 += qv * bf2f(row[256 + h]);
            s1 += qv * bf2f(row[256 + HID + h]);
        }
        const float scale = 0.0883883476483184f;   // 1/sqrt(128)
        s0 *= scale; s1 *= scale;
        float mx = fmaxf(s0, s1);
        float e0 = __expf(s0 - mx), e1 = __expf(s1 - mx);
        float inv = 1.f / (e0 + e1);
        float a0 = e0 * inv, a1 = e1 * inv;
        for (int h = 0; h < HID; ++h) {
            float v = a0 * bf2f(row[512 + h]) + a1 * bf2f(row[512 + HID + h]);
            smX[m * 256 + q * HID + h] = f32_to_bf16(v);
        }
    }
    __syncthreads();

    // build 8 A fragments from x = [16,256]
    Frag16 xa[8];
#pragma unroll
    for (int ks = 0; ks < 8; ++ks)
#pragma unroll
        for (int i = 0; i < 8; ++i) {
            int k = ks * 32 + 2 * kb + 2 * i;
            xa[ks].u[i] = *(const unsigned int*)&smX[m * KO + k];
        }
    __syncthreads();

    // ---- output projection: [16,256] x [256,128] + b_o
#pragma unroll 1
    for (int nt = 0; nt < 8; ++nt) {
        v8f acc = {0.f, 0.f, 0.f, 0.f, 0.f, 0.f, 0.f, 0.f};
#pragma unroll
        for (int ks = 0; ks < 8; ++ks) {
            Frag16 bf;
            const uint4* p = woFrag + (size_t)(ks * 8 + nt) * 64;
            bf.q[0] = p[2 * l];
            bf.q[1] = p[2 * l + 1];
            acc = __builtin_amdgcn_wmma_f32_16x16x32_bf16(false, xa[ks].v, false, bf.v,
                                                          (short)0, acc, false, false);
        }
        int col = nt * 16 + m;
        float bias = bo[col];
#pragma unroll
        for (int r = 0; r < 8; ++r) {
            int M = (l >> 4) * 8 + r;
            smXO[M * HID + col] = acc[r] + bias;
        }
    }
    __syncthreads();

    // ---- LN + residual, one atom per lane (lanes 0..15)
    if (l < 16) {
        float s = 0.f, ss = 0.f;
        for (int c = 0; c < HID; ++c) { float v = smXO[l * HID + c]; s += v; ss += v * v; }
        float mu  = s * (1.f / HID);
        float var = ss * (1.f / HID) - mu * mu;
        float inv = rsqrtf(var + 1e-5f);
        size_t base = (size_t)(tile * 16 + l) * HID;
        for (int c = 0; c < HID; ++c) {
            float v = smXO[l * HID + c];
            out[base + c] = smH[l * HID + c] + (g2[c] * (v - mu) * inv + b2[c]);
        }
    }
}

// ---------------------------------------------------------------------------
extern "C" void kernel_launch(void* const* d_in, const int* in_sizes, int n_in,
                              void* d_out, int out_size, void* d_ws, size_t ws_size,
                              hipStream_t stream) {
    (void)in_sizes; (void)n_in; (void)out_size; (void)ws_size;

    const float* fA = (const float*)d_in[0];
    const float* fB = (const float*)d_in[1];
    const int*   a2a = (const int*)d_in[2];
    const int*   a2b = (const int*)d_in[3];
    const float* Wi = (const float*)d_in[4];
    const float* bi = (const float*)d_in[5];
    const float* g1 = (const float*)d_in[6];
    const float* b1 = (const float*)d_in[7];
    const float* Wq = (const float*)d_in[8];
    const float* bq = (const float*)d_in[9];
    const float* Wk = (const float*)d_in[10];
    const float* bk = (const float*)d_in[11];
    const float* Wv = (const float*)d_in[12];
    const float* bv = (const float*)d_in[13];
    const float* Wo = (const float*)d_in[14];
    const float* bo = (const float*)d_in[15];
    const float* g2 = (const float*)d_in[16];
    const float* b2 = (const float*)d_in[17];

    char* ws = (char*)d_ws;
    float*          h0  = (float*)(ws + OFF_H0);
    unsigned short* msg = (unsigned short*)(ws + OFF_MSG);
    unsigned int*   wiF = (unsigned int*)(ws + OFF_WI);
    unsigned int*   whF = (unsigned int*)(ws + OFF_WH);
    unsigned int*   woF = (unsigned int*)(ws + OFF_WO);

    {   // weights -> bf16 fragment layouts (tiny, L2 resident thereafter)
        int t0 = 5 * 8 * 256;
        cvt_plain<<<(t0 + 255) / 256, 256, 0, stream>>>(Wi, wiF, AF, 8, t0);
        int t1 = 10 * 48 * 256;
        cvt_wh<<<(t1 + 255) / 256, 256, 0, stream>>>(Wq, Wk, Wv, whF, t1);
        int t2 = 8 * 8 * 256;
        cvt_plain<<<(t2 + 255) / 256, 256, 0, stream>>>(Wo, woF, KO, 8, t2);
    }

    k1_inproj<<<NATOMS / 16, 32, 0, stream>>>(fA, (const uint4*)wiF, bi, g1, b1, h0);
    k2_gather<<<NATOMS, 128, 0, stream>>>(h0, fB, a2a, a2b, msg);
    k3_fused<<<NATOMS / 16, 32, 0, stream>>>(msg, h0, (const uint4*)whF, (const uint4*)woF,
                                             bq, bk, bv, bo, g2, b2, (float*)d_out);
}
